// GlobalDownSample_68719477128
// MI455X (gfx1250) — matrix operand order repaired and checked by reading, not compiled, not run on hardware
//
#include <hip/hip_runtime.h>
#include <hip/hip_bf16.h>

// ---- problem constants (reference: B,C,N,M = 2,128,8192,2048) ----
#define BB 2
#define CC 128
#define NN 8192
#define MM 2048
#define NT (NN / 16)                // 512 tiles of 16 along N
#define INVS 0.08838834764831845f   // 1/sqrt(128)

typedef __attribute__((ext_vector_type(16))) __bf16 v16bf;
typedef __attribute__((ext_vector_type(8)))  float  v8f;

#if __has_builtin(__builtin_amdgcn_tensor_load_to_lds) && __has_builtin(__builtin_amdgcn_s_wait_tensorcnt)
#define HAVE_TDM 1
typedef __attribute__((ext_vector_type(4))) unsigned int v4u;
typedef __attribute__((ext_vector_type(8))) int          v8i;
typedef __attribute__((ext_vector_type(4))) int          v4i;
#else
#define HAVE_TDM 0
#endif

union BV {
    v16bf v;
    unsigned short h[16];
    unsigned int   w[8];
};

// Packed f32 -> 2 x bf16, round-to-nearest (ties up). One v_perm_b32 packs the
// two high halves; +0x8000 applies nearest rounding. (Shift/mask path, same rounding,
// as fallback only if v_perm builtin is unavailable.)
__device__ __forceinline__ unsigned int pack_bf16(float lo, float hi) {
    unsigned int ulo = __builtin_bit_cast(unsigned int, lo) + 0x8000u;
    unsigned int uhi = __builtin_bit_cast(unsigned int, hi) + 0x8000u;
#if __has_builtin(__builtin_amdgcn_perm)
    // result bytes: {uhi[3],uhi[2],ulo[3],ulo[2]} -> {bf16(hi), bf16(lo)}
    return __builtin_amdgcn_perm(uhi, ulo, 0x07060302u);
#else
    return (ulo >> 16) | (uhi & 0xFFFF0000u);
#endif
}

__device__ __forceinline__ unsigned short cvt_bf16(float x) {
    return (unsigned short)((__builtin_bit_cast(unsigned int, x) + 0x8000u) >> 16);
}

__device__ __forceinline__ v8f wmma_bf16(const BV& a, const BV& b, v8f c) {
    return __builtin_amdgcn_wmma_f32_16x16x32_bf16(
        false, a.v, false, b.v, (short)0, c, false, false);
}

// A-matrix (16x32 bf16, MxK) K-pair base for VGPR j, half h.
__device__ __forceinline__ int a_kbase(int j, int h) {
    return ((j < 4) ? (2 * j) : (16 + 2 * (j - 4))) + 8 * h;
}

// Gather A tile from bf16 matrix [row][k] (pairs contiguous -> u32 loads)
__device__ __forceinline__ void loadA_bf16(BV& a, const unsigned short* base,
                                           int stride, int row, int kofs, int h) {
#pragma unroll
    for (int j = 0; j < 8; ++j)
        a.w[j] = *(const unsigned int*)(base + row * stride + kofs + a_kbase(j, h));
}

__device__ __forceinline__ void loadA_f32(BV& a, const float* base,
                                          int stride, int row, int kofs, int h) {
#pragma unroll
    for (int j = 0; j < 8; ++j) {
        int kb = a_kbase(j, h);
        a.w[j] = pack_bf16(base[row * stride + kofs + kb],
                           base[row * stride + kofs + kb + 1]);
    }
}

// Gather B tile (32x16 bf16 KxN): VGPR j rows K = a_kbase(j,h), +1; source [k][col].
__device__ __forceinline__ void loadB_bf16(BV& b, const unsigned short* base,
                                           int stride, int krow0, int col, int h) {
#pragma unroll
    for (int j = 0; j < 8; ++j) {
        int k = krow0 + a_kbase(j, h);
        unsigned int lo = base[k * stride + col];
        unsigned int hi = base[(k + 1) * stride + col];
        b.w[j] = lo | (hi << 16);
    }
}

__device__ __forceinline__ void loadB_f32(BV& b, const float* base,
                                          int stride, int krow0, int col, int h) {
#pragma unroll
    for (int j = 0; j < 8; ++j) {
        int k = krow0 + a_kbase(j, h);
        b.w[j] = pack_bf16(base[k * stride + col], base[(k + 1) * stride + col]);
    }
}

__device__ __forceinline__ void bv_from2(BV& t, uint4 a, uint4 b) {
    t.w[0] = a.x; t.w[1] = a.y; t.w[2] = a.z; t.w[3] = a.w;
    t.w[4] = b.x; t.w[5] = b.y; t.w[6] = b.z; t.w[7] = b.w;
}

// Packed operand tile: 32 lanes x 8 dwords = 1 KiB, exactly the WMMA reg image.
// Hot-loop load = two b128 loads per operand.
__device__ __forceinline__ void loadPK(BV& t, const uint4* base, int lane) {
    bv_from2(t, base[lane * 2], base[lane * 2 + 1]);
}

#if HAVE_TDM
// Issue a 1D TDM copy (bytes multiple of 8) global -> LDS. D# per ISA 8.3-8.6:
// group0: count=1, lds_addr, global_addr, type=2("image").
// group1: data_size=3 (8B units), tensor_dim0=tile_dim0=units, dims1+ trivial.
// amdgpu-toolchain (clang-23) 6-arg form: (v4u, v8i, v4i, v4i, v8i, i32 cpol).
__device__ __forceinline__ void tdm_load_1d(unsigned int lds_off,
                                            const void* gsrc, int bytes) {
    unsigned long long ga = (unsigned long long)gsrc;
    int units = bytes >> 3;
    v4u g0;
    g0.x = 1u;                                   // count=1, user descriptor
    g0.y = lds_off;                              // lds_addr (bytes)
    g0.z = (unsigned int)ga;                     // global_addr[31:0]
    g0.w = (unsigned int)((ga >> 32) & 0x01FFFFFFu) | (2u << 30); // [56:32]+type
    v8i g1;
    g1[0] = 3 << 16;                             // data_size=8B, wg_mask=0
    g1[1] = (units & 0xFFFF) << 16;              // tensor_dim0 lo16
    g1[2] = ((units >> 16) & 0xFFFF) | (1 << 16);// tensor_dim0 hi16, tensor_dim1=1
    g1[3] = (units & 0xFFFF) << 16;              // tile_dim0 = units
    g1[4] = 1;                                   // tile_dim1=1, tile_dim2=0
    g1[5] = units;                               // tensor_dim0_stride lo32
    g1[6] = 0;
    g1[7] = 0;
    v4i z4 = {0, 0, 0, 0};
    v8i z8 = {0, 0, 0, 0, 0, 0, 0, 0};
    __builtin_amdgcn_tensor_load_to_lds(g0, g1, z4, z4, z8, 0);
}
#endif

// ---------------- kernel 1: QKV projection (f32 in, bf16 out) ----------------
// q -> qb[N][C] (A gather) and qcb[C][N] (B gather); k -> ktb[N][C] and
// kcb[C][N]; v -> vtb[N][C].
__global__ void k_qkv(const float* __restrict__ x,
                      const float* __restrict__ Wq,
                      const float* __restrict__ Wk,
                      const float* __restrict__ Wv,
                      unsigned short* __restrict__ qb,
                      unsigned short* __restrict__ qcb,
                      unsigned short* __restrict__ ktb,
                      unsigned short* __restrict__ kcb,
                      unsigned short* __restrict__ vtb) {
    int wid  = blockIdx.x * (blockDim.x >> 5) + (threadIdx.x >> 5);
    int lane = threadIdx.x & 31;
    int h = lane >> 4, lm = lane & 15;

    int nt = wid % NT;  int r1 = wid / NT;
    int ct = r1 & 7;    int r2 = r1 >> 3;
    int b  = r2 % BB;   int mat = r2 / BB;     // 0=q 1=k 2=v

    const float* W = (mat == 0) ? Wq : ((mat == 1) ? Wk : Wv);
    const float* xb = x + (size_t)b * CC * NN;

    v8f acc = {};
    BV a, bb;
#pragma unroll
    for (int ks = 0; ks < 4; ++ks) {
        loadA_f32(a, W, CC, ct * 16 + lm, ks * 32, h);
        loadB_f32(bb, xb, NN, ks * 32, nt * 16 + lm, h);
        acc = wmma_bf16(a, bb, acc);
    }
#pragma unroll
    for (int r = 0; r < 8; ++r) {
        int co = ct * 16 + r + 8 * h;
        int n  = nt * 16 + lm;
        unsigned short hv = cvt_bf16(acc[r]);
        size_t inc = (size_t)b * NN * CC + (size_t)n * CC + co;
        size_t icn = (size_t)b * CC * NN + (size_t)co * NN + n;
        if (mat == 0)      { qb[inc] = hv; qcb[icn] = hv; }
        else if (mat == 1) { ktb[inc] = hv; kcb[icn] = hv; }
        else               { vtb[inc] = hv; }
    }
}

// --------------- kernel 2: repack operands into WMMA register image ---------
// QA (A of q rows), KA (A of k rows), KB (B of k cols), VB (B of vT rows).
__global__ void k_pack(const unsigned short* __restrict__ qb,
                       const unsigned short* __restrict__ ktb,
                       const unsigned short* __restrict__ kcb,
                       const unsigned short* __restrict__ vtb,
                       uint4* __restrict__ QA, uint4* __restrict__ KA,
                       uint4* __restrict__ KB, uint4* __restrict__ VB) {
    int wid  = blockIdx.x * (blockDim.x >> 5) + (threadIdx.x >> 5);
    int lane = threadIdx.x & 31;
    int h = lane >> 4, lm = lane & 15;
    int arr = wid >> 12, rem = wid & 4095;

    BV t;
    uint4* dst;
    if (arr < 3) {
        int ks = rem & 3, tile = (rem >> 2) & (NT - 1), b = rem >> 11;
        size_t slot = ((size_t)(b * NT + tile) * 4 + ks) * 256;
        if (arr == 0) { loadA_bf16(t, qb  + (size_t)b * NN * CC, CC, tile * 16 + lm, ks * 32, h); dst = QA + slot; }
        else if (arr == 1) { loadA_bf16(t, ktb + (size_t)b * NN * CC, CC, tile * 16 + lm, ks * 32, h); dst = KA + slot; }
        else { loadB_bf16(t, kcb + (size_t)b * CC * NN, NN, ks * 32, tile * 16 + lm, h); dst = KB + slot; }
    } else {            // VB: tiles over K=n (32 per tile), col = c-tile
        int ct = rem & 7, nb = (rem >> 3) & 255, b = rem >> 11;
        loadB_bf16(t, vtb + (size_t)b * NN * CC + (size_t)nb * 32 * CC, CC, 0, ct * 16 + lm, h);
        dst = VB + ((size_t)(b * (NN / 32) + nb) * 8 + ct) * 256;
    }
    dst[lane * 2]     = make_uint4(t.w[0], t.w[1], t.w[2], t.w[3]);
    dst[lane * 2 + 1] = make_uint4(t.w[4], t.w[5], t.w[6], t.w[7]);
}

// ------- kernel 3: row max / 1/rowsum via energy^T (per-lane softmax) --------
// E^T[m,n]: A = k rows (streamed, packed), B = q cols (fixed, preloaded).
// Softmax row index n is the D-tile COLUMN -> per-lane running max/sum.
__global__ void k_rowstats(const uint4* __restrict__ KA,
                           const unsigned short* __restrict__ qcb,
                           float* __restrict__ rowmax,
                           float* __restrict__ rowsuminv) {
    int wid  = blockIdx.x * (blockDim.x >> 5) + (threadIdx.x >> 5);
    int lane = threadIdx.x & 31;
    int h = lane >> 4, lm = lane & 15;
    int nblk = wid % NT, b = wid / NT;

    BV qv[4];
#pragma unroll
    for (int ks = 0; ks < 4; ++ks)
        loadB_bf16(qv[ks], qcb + (size_t)b * CC * NN, NN, ks * 32, nblk * 16 + lm, h);

    float runmax = -__builtin_inff(), runsum = 0.f;
    const uint4* kab = KA + (size_t)b * NT * 4 * 256;

    for (int mt = 0; mt < NT; ++mt) {
        __builtin_prefetch(kab + (size_t)((mt + 1) & (NT - 1)) * 4 * 256 + lane * 2, 0, 1);
        v8f acc = {};
#pragma unroll
        for (int ks = 0; ks < 4; ++ks) {
            BV a;
            loadPK(a, kab + (size_t)(mt * 4 + ks) * 256, lane);
            acc = wmma_bf16(a, qv[ks], acc);
        }
        float tmax = acc[0] * INVS;
#pragma unroll
        for (int r = 1; r < 8; ++r) tmax = fmaxf(tmax, acc[r] * INVS);
        float nm = fmaxf(runmax, tmax);
        float s = 0.f;
#pragma unroll
        for (int r = 0; r < 8; ++r) s += __expf(acc[r] * INVS - nm);
        runsum = runsum * __expf(runmax - nm) + s;
        runmax = nm;
    }
    // combine the two half-wave partials (rows split across halves)
    float rmO = __shfl_xor(runmax, 16, 32);
    float rsO = __shfl_xor(runsum, 16, 32);
    float nm  = fmaxf(runmax, rmO);
    float tot = runsum * __expf(runmax - nm) + rsO * __expf(rmO - nm);
    if (lane < 16) {
        int n = nblk * 16 + lane;
        rowmax[b * NN + n]    = nm;
        rowsuminv[b * NN + n] = 1.0f / tot;
    }
}

// ------------- kernel 4: column sums of attention (selection scores) ---------
__global__ void k_colsum(const uint4* __restrict__ QA,
                         const unsigned short* __restrict__ kcb,
                         const float* __restrict__ rowmax,
                         const float* __restrict__ rowsuminv,
                         float* __restrict__ colsum) {
    int wid  = blockIdx.x * (blockDim.x >> 5) + (threadIdx.x >> 5);
    int lane = threadIdx.x & 31;
    int h = lane >> 4, lm = lane & 15;
    int mblk = wid % NT, b = wid / NT;

    BV kv[4];
#pragma unroll
    for (int ks = 0; ks < 4; ++ks)
        loadB_bf16(kv[ks], kcb + (size_t)b * CC * NN, NN, ks * 32, mblk * 16 + lm, h);

    const uint4* qab = QA + (size_t)b * NT * 4 * 256;
    float csum = 0.f;
    for (int nt = 0; nt < NT; ++nt) {
        __builtin_prefetch(qab + (size_t)((nt + 1) & (NT - 1)) * 4 * 256 + lane * 2, 0, 1);
        v8f acc = {};
#pragma unroll
        for (int ks = 0; ks < 4; ++ks) {
            BV a;
            loadPK(a, qab + (size_t)(nt * 4 + ks) * 256, lane);
            acc = wmma_bf16(a, kv[ks], acc);
        }
#pragma unroll
        for (int r = 0; r < 8; ++r) {
            int n = nt * 16 + r + 8 * h;
            csum += __expf(acc[r] * INVS - rowmax[b * NN + n]) * rowsuminv[b * NN + n];
        }
    }
    csum += __shfl_xor(csum, 16, 32);
    if (lane < 16) colsum[b * NN + mblk * 16 + lane] = csum;
}

// ------------- kernel 5: exact top-M by stable rank (matches lax.top_k) ------
__global__ void k_topk(const float* __restrict__ colsum, int* __restrict__ idxo) {
    int t = threadIdx.x;
    int gid = blockIdx.x * 256 + t;
    int b = gid / NN, n = gid % NN;
    float myv = colsum[b * NN + n];
    __shared__ __align__(16) float sv[256];
    int cnt = 0;
    for (int c0 = 0; c0 < NN; c0 += 256) {
        __syncthreads();
        sv[t] = colsum[b * NN + c0 + t];
        __syncthreads();
#pragma unroll 8
        for (int j = 0; j < 256; ++j) {
            float o = sv[j]; int jg = c0 + j;
            cnt += (o > myv) || (o == myv && jg < n);
        }
    }
    if (cnt < MM) idxo[b * MM + cnt] = n;
}

// ------------- kernel 6: gather selected rows, scores GEMM with v ------------
// 4 waves / block. Dynamic LDS per wave: 2 x 8KiB TDM stage + 1KiB score tile.
__global__ void k_out(const unsigned short* __restrict__ qb,
                      const uint4* __restrict__ KB,
                      const uint4* __restrict__ VB,
                      const float* __restrict__ rowmax,
                      const float* __restrict__ rowsuminv,
                      const int* __restrict__ idxo,
                      float* __restrict__ out) {
    extern __shared__ char smem[];
    int w    = threadIdx.x >> 5;
    int wid  = blockIdx.x * (blockDim.x >> 5) + w;
    int lane = threadIdx.x & 31;
    int h = lane >> 4, lm = lane & 15;
    int mblk = wid % (MM / 16), b = wid / (MM / 16);
    int m0 = mblk * 16;

    char* kstage = smem + w * 16384;                        // 2 x 8 KiB
    unsigned short(*sS)[32] = (unsigned short(*)[32])(smem + 4 * 16384 + w * 1024);

    // gathered q rows (A layout), per-row softmax stats
    int nrowA = idxo[b * MM + m0 + lm];
    BV qa[4];
#pragma unroll
    for (int ks = 0; ks < 4; ++ks)
        loadA_bf16(qa[ks], qb + (size_t)b * NN * CC, CC, nrowA, ks * 32, h);
    float rm[8], ri[8];
#pragma unroll
    for (int r = 0; r < 8; ++r) {
        int nr = idxo[b * MM + m0 + r + 8 * h];
        rm[r] = rowmax[b * NN + nr];
        ri[r] = rowsuminv[b * NN + nr];
    }

    v8f acc[8];
#pragma unroll
    for (int ct = 0; ct < 8; ++ct) acc[ct] = {};

    const uint4* kbb = KB + (size_t)b * NT * 4 * 256;       // 8 KiB per n0 step
    const uint4* vbb = VB + (size_t)b * (NN / 32) * 8 * 256;

#if HAVE_TDM
    tdm_load_1d((unsigned int)(size_t)kstage, kbb, 8192);   // prologue: buf 0
#endif

    for (int n0 = 0; n0 < NN; n0 += 32) {
        int cur = (n0 >> 5) & 1;
#if HAVE_TDM
        if (n0 + 32 < NN)
            tdm_load_1d((unsigned int)(size_t)(kstage + (cur ^ 1) * 8192),
                        kbb + (size_t)((n0 >> 4) + 2) * 4 * 256, 8192);
        if (n0 + 32 < NN) __builtin_amdgcn_s_wait_tensorcnt(1);
        else              __builtin_amdgcn_s_wait_tensorcnt(0);
#endif
        // 16x32 score block: two 16x16 energy tiles, normalized, staged to LDS
#pragma unroll
        for (int t = 0; t < 2; ++t) {
            v8f e = {};
#pragma unroll
            for (int ks = 0; ks < 4; ++ks) {
                BV bb;
#if HAVE_TDM
                loadPK(bb, (const uint4*)(kstage + cur * 8192 + (t * 4 + ks) * 1024), lane);
#else
                loadPK(bb, kbb + (size_t)(((n0 >> 4) + t) * 4 + ks) * 256, lane);
#endif
                e = wmma_bf16(qa[ks], bb, e);
            }
#pragma unroll
            for (int r = 0; r < 8; ++r)
                sS[r + 8 * h][16 * t + lm] = cvt_bf16(__expf(e[r] * INVS - rm[r]) * ri[r]);
        }
        __syncthreads();
        BV sa;          // scores re-read in A layout (K = 32 staged columns)
#pragma unroll
        for (int j = 0; j < 8; ++j)
            sa.w[j] = *(const unsigned int*)&sS[lm][a_kbase(j, h)];
        __builtin_prefetch(vbb + (size_t)((n0 >> 5) + 1) * 8 * 256 + lane * 2, 0, 1);
#pragma unroll
        for (int ct = 0; ct < 8; ++ct) {
            BV bv;
            loadPK(bv, vbb + (size_t)((n0 >> 5) * 8 + ct) * 256, lane);
            acc[ct] = wmma_bf16(sa, bv, acc[ct]);
        }
        __syncthreads();
    }
#pragma unroll
    for (int ct = 0; ct < 8; ++ct)
#pragma unroll
        for (int r = 0; r < 8; ++r)
            out[((size_t)b * CC + ct * 16 + lm) * MM + m0 + r + 8 * h] = acc[ct][r];
}

extern "C" void kernel_launch(void* const* d_in, const int* in_sizes, int n_in,
                              void* d_out, int out_size, void* d_ws, size_t ws_size,
                              hipStream_t stream) {
    (void)in_sizes; (void)n_in; (void)out_size; (void)ws_size;
    const float* x  = (const float*)d_in[0];
    const float* Wq = (const float*)d_in[1];
    const float* Wk = (const float*)d_in[2];
    const float* Wv = (const float*)d_in[3];
    float* out = (float*)d_out;

    char* ws = (char*)d_ws;
    size_t o = 0;
    const size_t PLN = (size_t)BB * NN * CC * 2;     // 4 MiB each
    unsigned short* qb  = (unsigned short*)(ws + o); o += PLN;
    unsigned short* qcb = (unsigned short*)(ws + o); o += PLN;
    unsigned short* ktb = (unsigned short*)(ws + o); o += PLN;
    unsigned short* kcb = (unsigned short*)(ws + o); o += PLN;
    unsigned short* vtb = (unsigned short*)(ws + o); o += PLN;
    const size_t PKN = (size_t)BB * NT * 4 * 256 * 16;   // 16 MiB each (packed)
    uint4* QA = (uint4*)(ws + o); o += PKN;
    uint4* KA = (uint4*)(ws + o); o += PKN;
    uint4* KB = (uint4*)(ws + o); o += PKN;
    uint4* VB = (uint4*)(ws + o); o += PKN;
    float* rowmax    = (float*)(ws + o); o += (size_t)BB * NN * 4;
    float* rowsuminv = (float*)(ws + o); o += (size_t)BB * NN * 4;
    float* colsum    = (float*)(ws + o); o += (size_t)BB * NN * 4;
    int*   idxo      = (int*)(ws + o);   o += (size_t)BB * MM * 4;

    dim3 blk(256);
    k_qkv<<<3072, blk, 0, stream>>>(x, Wq, Wk, Wv, qb, qcb, ktb, kcb, vtb);
    k_pack<<<2048, blk, 0, stream>>>(qb, ktb, kcb, vtb, QA, KA, KB, VB);
    k_rowstats<<<128, blk, 0, stream>>>(KA, qcb, rowmax, rowsuminv);
    k_colsum<<<128, blk, 0, stream>>>(QA, kcb, rowmax, rowsuminv, colsum);
    k_topk<<<64, blk, 0, stream>>>(colsum, idxo);
    // 4 waves/block, dynamic LDS: 4*(2*8KiB) TDM stage + 4*1KiB score tiles
    k_out<<<64, dim3(128), 4 * 16384 + 4 * 1024, stream>>>(
        qb, KB, VB, rowmax, rowsuminv, idxo, out);
}
